// GraphEncoderLayer_40183714021846
// MI455X (gfx1250) — matrix operand order repaired and checked by reading, not compiled, not run on hardware
//
#include <hip/hip_runtime.h>

// GraphEncoderLayer for MI455X (gfx1250): bf16 WMMA GEMMs + flash attention,
// double-buffered async global->LDS staging (GLOBAL_LOAD_ASYNC_TO_LDS_B128).
// B=16, L=512, D=512, H=8, hd=64, FF=2048, RB=16.

#define B_  16
#define L_  512
#define D_  512
#define H_  8
#define HD_ 64
#define FF_ 2048
#define RB_ 16
#define EPS_ 1e-5f

typedef __attribute__((ext_vector_type(16))) __bf16 v16bf;
typedef __attribute__((ext_vector_type(8)))  float v8f;

union Frag16 {
  v16bf bf;
  unsigned short us[16];
  uint4 u4[2];
};

__device__ __forceinline__ unsigned short f2bf(float f) {
  unsigned int u = __float_as_uint(f);
  u += 0x7FFFu + ((u >> 16) & 1u);   // round-to-nearest-even
  return (unsigned short)(u >> 16);
}
__device__ __forceinline__ float bf2f(unsigned short h) {
  return __uint_as_float(((unsigned int)h) << 16);
}
__device__ __forceinline__ v8f wmma_bf16(const Frag16& a, const Frag16& b, v8f c) {
  return __builtin_amdgcn_wmma_f32_16x16x32_bf16(false, a.bf, false, b.bf,
                                                 (short)0, c, false, false);
}
__device__ __forceinline__ float hmax16(float x) {
  #pragma unroll
  for (int m = 1; m < 16; m <<= 1) x = fmaxf(x, __shfl_xor(x, m, 32));
  return x;
}
__device__ __forceinline__ float hsum16(float x) {
  #pragma unroll
  for (int m = 1; m < 16; m <<= 1) x += __shfl_xor(x, m, 32);
  return x;
}
__device__ __forceinline__ uint2 load4bf(const float* p) {
  const float4 f = *(const float4*)p;
  uint2 r;
  r.x = (unsigned)f2bf(f.x) | ((unsigned)f2bf(f.y) << 16);
  r.y = (unsigned)f2bf(f.z) | ((unsigned)f2bf(f.w) << 16);
  return r;
}

// gfx1250 async global->LDS copy, 16B per lane; tracked by ASYNCcnt.
__device__ __forceinline__ void async_b128(unsigned lds_off, const void* gaddr) {
  asm volatile("global_load_async_to_lds_b128 %0, %1, off"
               :: "v"(lds_off), "v"((unsigned long long)gaddr) : "memory");
}
__device__ __forceinline__ void wait_async0() {
  asm volatile("s_wait_asynccnt 0x0" ::: "memory");
}
__device__ __forceinline__ unsigned ldsOff(const void* p) {
  return (unsigned)(size_t)p;   // generic shared ptr: addr[31:0] = LDS offset
}

// ---------------------------------------------------------------------------
// GEMM: C[M,N] = A[M,K] @ W[K,N] (+bias, relu, +residual).
// A bf16 [M,K]; W pre-transposed bf16 Wt[N,K]. BM=128 BN=128 BK=32.
// 256 threads = 8 waves (4x2); wave tile 32x64 (8 accum frags, 8 WMMA/step).
// Double-buffered LDS: async-stage tile i+1 while WMMAs consume tile i.
// ---------------------------------------------------------------------------
__global__ void __launch_bounds__(256)
gemm_kernel(const unsigned short* __restrict__ A, const unsigned short* __restrict__ Wt,
            const float* __restrict__ bias, const float* __restrict__ residual,
            float* __restrict__ outF, unsigned short* __restrict__ outBf,
            int M, int K, int N, int relu) {
  __shared__ unsigned short sA [2][128][40];  // [m][k], 80B stride
  __shared__ unsigned short sBt[2][128][40];  // [n][k]

  const int bm = blockIdx.y * 128;
  const int bn = blockIdx.x * 128;
  const int tid  = threadIdx.x;
  const int wave = tid >> 5, lane = tid & 31;
  const int half = lane >> 4, ln = lane & 15;
  const int wr = wave >> 1, wc = wave & 1;

  // Per-thread staging coordinates: 512 b128 units per tile -> 2 per thread.
  const int sr = tid >> 2, sc8 = (tid & 3) * 8;

  auto stage = [&](int buf, int k0) {
    async_b128(ldsOff(&sA [buf][sr][sc8]),      A  + (size_t)(bm + sr)      * K + (k0 + sc8));
    async_b128(ldsOff(&sBt[buf][sr][sc8]),      Wt + (size_t)(bn + sr)      * K + (k0 + sc8));
    async_b128(ldsOff(&sA [buf][sr + 64][sc8]), A  + (size_t)(bm + sr + 64) * K + (k0 + sc8));
    async_b128(ldsOff(&sBt[buf][sr + 64][sc8]), Wt + (size_t)(bn + sr + 64) * K + (k0 + sc8));
  };

  v8f acc[2][4] = {};

  stage(0, 0);
  wait_async0();
  __syncthreads();

  int cur = 0;
  for (int k0 = 0; k0 < K; k0 += 32) {
    // Kick off next tile into the alternate buffer; overlap with WMMA below.
    if (k0 + 32 < K) stage(cur ^ 1, k0 + 32);
    if (k0 + 64 < K) {
      __builtin_prefetch(A  + (size_t)(bm + (tid & 127)) * K + (k0 + 64), 0, 3);
      __builtin_prefetch(Wt + (size_t)(bn + (tid & 127)) * K + (k0 + 64), 0, 3);
    }

    Frag16 af[2], bfg[4];
    #pragma unroll
    for (int mi = 0; mi < 2; mi++) {
      int row = wr * 32 + mi * 16 + ln;
      af[mi].u4[0] = *(const uint4*)&sA[cur][row][8 * half];        // K = 8h+0..7
      af[mi].u4[1] = *(const uint4*)&sA[cur][row][16 + 8 * half];   // K = 16+8h+0..7
    }
    #pragma unroll
    for (int ni = 0; ni < 4; ni++) {
      int col = wc * 64 + ni * 16 + ln;
      bfg[ni].u4[0] = *(const uint4*)&sBt[cur][col][8 * half];      // K = 8h+0..7
      bfg[ni].u4[1] = *(const uint4*)&sBt[cur][col][16 + 8 * half]; // K = 16+8h+0..7
    }
    #pragma unroll
    for (int mi = 0; mi < 2; mi++)
      #pragma unroll
      for (int ni = 0; ni < 4; ni++)
        acc[mi][ni] = wmma_bf16(af[mi], bfg[ni], acc[mi][ni]);

    wait_async0();
    __syncthreads();
    cur ^= 1;
  }

  #pragma unroll
  for (int mi = 0; mi < 2; mi++)
    #pragma unroll
    for (int ni = 0; ni < 4; ni++)
      #pragma unroll
      for (int r = 0; r < 8; r++) {
        int gm = bm + wr * 32 + mi * 16 + 8 * half + r;
        int gn = bn + wc * 64 + ni * 16 + ln;
        float v = acc[mi][ni][r];
        if (bias)     v += bias[gn];
        if (relu)     v  = fmaxf(v, 0.0f);
        if (residual) v += residual[(size_t)gm * N + gn];
        if (outF)  outF [(size_t)gm * N + gn] = v;
        if (outBf) outBf[(size_t)gm * N + gn] = f2bf(v);
      }
}

// ---------------------------------------------------------------------------
// One-time prep: fp32 -> bf16 elementwise, and fp32 [K,N] -> bf16 Wt[N,K].
// ---------------------------------------------------------------------------
__global__ void __launch_bounds__(256)
cvtbf_kernel(const float* __restrict__ x, unsigned short* __restrict__ y) {
  size_t i = ((size_t)blockIdx.x * 256 + threadIdx.x) * 4;
  *(uint2*)(y + i) = load4bf(x + i);
}

__global__ void __launch_bounds__(256)
transpose_bf_kernel(const float* __restrict__ W, unsigned short* __restrict__ Wt,
                    int K, int N) {
  __shared__ float tile[32][33];
  const int bx = blockIdx.x * 32;  // n
  const int by = blockIdx.y * 32;  // k
  const int tx = threadIdx.x & 31, ty = threadIdx.x >> 5;
  #pragma unroll
  for (int yy = ty; yy < 32; yy += 8)
    tile[yy][tx] = W[(size_t)(by + yy) * N + (bx + tx)];
  __syncthreads();
  #pragma unroll
  for (int yy = ty; yy < 32; yy += 8)
    Wt[(size_t)(bx + yy) * K + (by + tx)] = f2bf(tile[tx][yy]);
}

// ---------------------------------------------------------------------------
// Edge-bias: rbf(cost) -> relu(@We1+be1) -> @We2+be2, stored bf16 [B,H,L,L].
// ---------------------------------------------------------------------------
__global__ void __launch_bounds__(256)
edge_kernel(const float* __restrict__ cost, const float* __restrict__ We1,
            const float* __restrict__ be1, const float* __restrict__ We2,
            const float* __restrict__ be2, unsigned short* __restrict__ biasOut) {
  __shared__ float sW1[RB_ * RB_], sb1[RB_], sW2[RB_ * H_], sb2[H_];
  const int tid = threadIdx.x;
  if (tid < RB_ * RB_) sW1[tid] = We1[tid];
  if (tid < RB_)       sb1[tid] = be1[tid];
  if (tid < RB_ * H_)  sW2[tid] = We2[tid];
  if (tid < H_)        sb2[tid] = be2[tid];
  __syncthreads();

  const size_t gid = (size_t)blockIdx.x * 256 + tid;
  const int j = (int)(gid % L_);
  const size_t t = gid / L_;
  const int i = (int)(t % L_);
  const int b = (int)(t / L_);

  const float c = cost[gid];
  float rbf[RB_];
  #pragma unroll
  for (int r = 0; r < RB_; r++) {
    float d = c - (float)r * (1.0f / 15.0f);
    rbf[r] = __expf(-d * d * 112.5f);
  }
  float hid[RB_];
  #pragma unroll
  for (int u = 0; u < RB_; u++) {
    float s = sb1[u];
    #pragma unroll
    for (int r = 0; r < RB_; r++) s += rbf[r] * sW1[r * RB_ + u];
    hid[u] = fmaxf(s, 0.0f);
  }
  #pragma unroll
  for (int hh = 0; hh < H_; hh++) {
    float s = sb2[hh];
    #pragma unroll
    for (int u = 0; u < RB_; u++) s += hid[u] * sW2[u * H_ + hh];
    biasOut[(((size_t)(b * H_ + hh) * L_ + i) * L_) + j] = f2bf(s);
  }
}

// ---------------------------------------------------------------------------
// Flash attention with additive bias. Block = (b, h, 128 query rows), 8 waves
// of 16 rows each; j-tiles of 32 keys, double-buffered: K & bias async-staged,
// V staged transposed, all into the alternate buffer while WMMAs consume the
// current one. Output written as bf16 (A of the Wo GEMM).
// ---------------------------------------------------------------------------
__global__ void __launch_bounds__(256)
attn_kernel(const unsigned short* __restrict__ qg, const unsigned short* __restrict__ kg,
            const unsigned short* __restrict__ vg, const unsigned short* __restrict__ biasg,
            unsigned short* __restrict__ ctxbf) {
  const int iblk = blockIdx.x;
  const int h    = blockIdx.y;
  const int b    = blockIdx.z;
  const int tid  = threadIdx.x;
  const int wave = tid >> 5, lane = tid & 31;
  const int half = lane >> 4, ln = lane & 15;

  __shared__ unsigned short sK   [2][32][72];     // [key][hd], 144B stride
  __shared__ unsigned short sVt  [2][64][40];     // [hd][key]
  __shared__ unsigned short sBias[2][128][40];    // 80B stride
  __shared__ unsigned short sP   [8][16][32];     // per-wave P (C->A relayout)

  const int qbase = iblk * 128 + wave * 16;
  const size_t biasRowBase = (size_t)(b * H_ + h) * L_;

  auto stage = [&](int buf, int j0) {
    // K tile: 32x64 = 256 b128 units, one per thread.
    {
      int r = tid >> 3, c8 = (tid & 7) * 8;
      async_b128(ldsOff(&sK[buf][r][c8]),
                 kg + (size_t)(b * L_ + j0 + r) * D_ + h * HD_ + c8);
    }
    // Bias tile: 128x32 = 512 units, two per thread.
    {
      int r = tid >> 2, c8 = (tid & 3) * 8;
      async_b128(ldsOff(&sBias[buf][r][c8]),
                 biasg + (biasRowBase + iblk * 128 + r) * L_ + j0 + c8);
      async_b128(ldsOff(&sBias[buf][r + 64][c8]),
                 biasg + (biasRowBase + iblk * 128 + r + 64) * L_ + j0 + c8);
    }
    // V transposed: uint2 reads, u16 scatter (synchronous).
    #pragma unroll
    for (int t = tid; t < 32 * 64 / 4; t += 256) {
      int r = t >> 4, c4 = (t & 15) * 4;
      uint2 vv = *(const uint2*)(vg + (size_t)(b * L_ + j0 + r) * D_ + h * HD_ + c4);
      sVt[buf][c4 + 0][r] = (unsigned short)(vv.x);
      sVt[buf][c4 + 1][r] = (unsigned short)(vv.x >> 16);
      sVt[buf][c4 + 2][r] = (unsigned short)(vv.y);
      sVt[buf][c4 + 3][r] = (unsigned short)(vv.y >> 16);
    }
  };

  Frag16 qf[2];
  const unsigned short* qrow =
      qg + ((size_t)(b * L_ + qbase + ln) * D_ + h * HD_);
  #pragma unroll
  for (int c = 0; c < 2; c++) {
    qf[c].u4[0] = *(const uint4*)(qrow + c * 32 + 8 * half);
    qf[c].u4[1] = *(const uint4*)(qrow + c * 32 + 16 + 8 * half);
  }

  float mi[8], li[8];
  #pragma unroll
  for (int r = 0; r < 8; r++) { mi[r] = -3.0e38f; li[r] = 0.0f; }
  v8f o0 = {}, o1 = {}, o2 = {}, o3 = {};

  stage(0, 0);
  wait_async0();
  __syncthreads();

  int cur = 0;
  for (int j0 = 0; j0 < L_; j0 += 32) {
    if (j0 + 32 < L_) stage(cur ^ 1, j0 + 32);

    // scores: Q[16,64] @ K^T[64,32] as two 16-key fragments
    v8f s0 = {}, s1 = {};
    #pragma unroll
    for (int c = 0; c < 2; c++) {
      Frag16 kb0, kb1;
      kb0.u4[0] = *(const uint4*)&sK[cur][ln][c * 32 + 8 * half];
      kb0.u4[1] = *(const uint4*)&sK[cur][ln][c * 32 + 16 + 8 * half];
      kb1.u4[0] = *(const uint4*)&sK[cur][16 + ln][c * 32 + 8 * half];
      kb1.u4[1] = *(const uint4*)&sK[cur][16 + ln][c * 32 + 16 + 8 * half];
      s0 = wmma_bf16(qf[c], kb0, s0);
      s1 = wmma_bf16(qf[c], kb1, s1);
    }

    // bias add + online softmax per accumulator row (m = 8*half + r)
    #pragma unroll
    for (int r = 0; r < 8; r++) {
      int mrow = wave * 16 + 8 * half + r;
      float v0 = s0[r] * 0.125f + bf2f(sBias[cur][mrow][ln]);
      float v1 = s1[r] * 0.125f + bf2f(sBias[cur][mrow][16 + ln]);
      float tm = hmax16(fmaxf(v0, v1));
      float nm = fmaxf(mi[r], tm);
      float sc = __expf(mi[r] - nm);
      float p0 = __expf(v0 - nm);
      float p1 = __expf(v1 - nm);
      li[r] = li[r] * sc + hsum16(p0 + p1);
      mi[r] = nm;
      o0[r] *= sc; o1[r] *= sc; o2[r] *= sc; o3[r] *= sc;
      sP[wave][8 * half + r][ln]      = f2bf(p0);
      sP[wave][8 * half + r][16 + ln] = f2bf(p1);
    }

    // P (16x32) @ V (32x64)
    Frag16 pa;
    pa.u4[0] = *(const uint4*)&sP[wave][ln][8 * half];
    pa.u4[1] = *(const uint4*)&sP[wave][ln][16 + 8 * half];
    #pragma unroll
    for (int nc = 0; nc < 4; nc++) {
      Frag16 vb;
      vb.u4[0] = *(const uint4*)&sVt[cur][nc * 16 + ln][8 * half];
      vb.u4[1] = *(const uint4*)&sVt[cur][nc * 16 + ln][16 + 8 * half];
      if      (nc == 0) o0 = wmma_bf16(pa, vb, o0);
      else if (nc == 1) o1 = wmma_bf16(pa, vb, o1);
      else if (nc == 2) o2 = wmma_bf16(pa, vb, o2);
      else              o3 = wmma_bf16(pa, vb, o3);
    }

    wait_async0();
    __syncthreads();
    cur ^= 1;
  }

  #pragma unroll
  for (int r = 0; r < 8; r++) {
    float inv = 1.0f / li[r];
    size_t row = (size_t)(b * L_ + qbase + 8 * half + r) * D_ + h * HD_;
    ctxbf[row +  0 + ln] = f2bf(o0[r] * inv);
    ctxbf[row + 16 + ln] = f2bf(o1[r] * inv);
    ctxbf[row + 32 + ln] = f2bf(o2[r] * inv);
    ctxbf[row + 48 + ln] = f2bf(o3[r] * inv);
  }
}

// ---------------------------------------------------------------------------
// In-place LayerNorm over D=512 per row; optional bf16 copy of the output.
// ---------------------------------------------------------------------------
__global__ void __launch_bounds__(256)
ln_kernel(float* __restrict__ x, const float* __restrict__ g,
          const float* __restrict__ bb, unsigned short* __restrict__ ybf) {
  __shared__ float red[256];
  const int tid = threadIdx.x;
  float* row = x + (size_t)blockIdx.x * D_;
  float a = row[tid], c = row[tid + 256];

  red[tid] = a + c;
  __syncthreads();
  #pragma unroll
  for (int s = 128; s > 0; s >>= 1) {
    if (tid < s) red[tid] += red[tid + s];
    __syncthreads();
  }
  float mean = red[0] * (1.0f / D_);
  __syncthreads();

  float da = a - mean, dc = c - mean;
  red[tid] = da * da + dc * dc;
  __syncthreads();
  #pragma unroll
  for (int s = 128; s > 0; s >>= 1) {
    if (tid < s) red[tid] += red[tid + s];
    __syncthreads();
  }
  float rs = __frsqrt_rn(red[0] * (1.0f / D_) + EPS_);

  float r0 = da * rs * g[tid]       + bb[tid];
  float r1 = dc * rs * g[tid + 256] + bb[tid + 256];
  row[tid]       = r0;
  row[tid + 256] = r1;
  if (ybf) {
    ybf[(size_t)blockIdx.x * D_ + tid]       = f2bf(r0);
    ybf[(size_t)blockIdx.x * D_ + tid + 256] = f2bf(r1);
  }
}

// ---------------------------------------------------------------------------
extern "C" void kernel_launch(void* const* d_in, const int* in_sizes, int n_in,
                              void* d_out, int out_size, void* d_ws, size_t ws_size,
                              hipStream_t stream) {
  const float* h_in = (const float*)d_in[0];
  const float* cost = (const float*)d_in[1];
  const float* Wq   = (const float*)d_in[2];
  const float* Wk   = (const float*)d_in[3];
  const float* Wv   = (const float*)d_in[4];
  const float* Wo   = (const float*)d_in[5];
  const float* We1  = (const float*)d_in[6];
  const float* be1  = (const float*)d_in[7];
  const float* We2  = (const float*)d_in[8];
  const float* be2  = (const float*)d_in[9];
  const float* g1   = (const float*)d_in[10];
  const float* bn1  = (const float*)d_in[11];
  const float* Wf1  = (const float*)d_in[12];
  const float* bf1  = (const float*)d_in[13];
  const float* Wf2  = (const float*)d_in[14];
  const float* bf2  = (const float*)d_in[15];
  const float* g2   = (const float*)d_in[16];
  const float* bn2  = (const float*)d_in[17];
  float* out = (float*)d_out;

  const size_t M  = (size_t)B_ * L_;          // 8192
  const size_t MD = M * D_;

  unsigned short* p = (unsigned short*)d_ws;
  unsigned short* q     = p; p += MD;                       // 8MB
  unsigned short* k     = p; p += MD;                       // 8MB
  unsigned short* v     = p; p += MD;                       // 8MB
  unsigned short* ebias = p; p += (size_t)B_ * H_ * L_ * L_;// 64MB
  unsigned short* ctxbf = p; p += MD;                       // 8MB
  unsigned short* hbf   = p; p += MD;                       // 8MB (bf16 of h_in)
  unsigned short* hbf2  = p; p += MD;                       // 8MB (bf16 of LN1)
  unsigned short* ff1   = p; p += M * FF_;                  // 32MB
  unsigned short* wtq   = p; p += (size_t)D_ * D_;          // 0.5MB each
  unsigned short* wtk   = p; p += (size_t)D_ * D_;
  unsigned short* wtv   = p; p += (size_t)D_ * D_;
  unsigned short* wto   = p; p += (size_t)D_ * D_;
  unsigned short* wtf1  = p; p += (size_t)D_ * FF_;         // 2MB
  unsigned short* wtf2  = p; p += (size_t)D_ * FF_;         // 2MB
  float* hbuf = (float*)p;                                  // 16MB fp32

  dim3 blk(256);
  dim3 gD(D_ / 128, (unsigned)(M / 128));     // (4, 64)
  dim3 gF(FF_ / 128, (unsigned)(M / 128));    // (16, 64)

  // One-time prep: bf16 activations + transposed bf16 weights.
  cvtbf_kernel<<<(unsigned)(MD / 1024), blk, 0, stream>>>(h_in, hbf);
  transpose_bf_kernel<<<dim3(D_/32,  D_/32),  blk, 0, stream>>>(Wq,  wtq,  D_,  D_);
  transpose_bf_kernel<<<dim3(D_/32,  D_/32),  blk, 0, stream>>>(Wk,  wtk,  D_,  D_);
  transpose_bf_kernel<<<dim3(D_/32,  D_/32),  blk, 0, stream>>>(Wv,  wtv,  D_,  D_);
  transpose_bf_kernel<<<dim3(D_/32,  D_/32),  blk, 0, stream>>>(Wo,  wto,  D_,  D_);
  transpose_bf_kernel<<<dim3(FF_/32, D_/32),  blk, 0, stream>>>(Wf1, wtf1, D_,  FF_);
  transpose_bf_kernel<<<dim3(D_/32,  FF_/32), blk, 0, stream>>>(Wf2, wtf2, FF_, D_);

  // QKV projections -> bf16
  gemm_kernel<<<gD, blk, 0, stream>>>(hbf, wtq, nullptr, nullptr, nullptr, q, (int)M, D_, D_, 0);
  gemm_kernel<<<gD, blk, 0, stream>>>(hbf, wtk, nullptr, nullptr, nullptr, k, (int)M, D_, D_, 0);
  gemm_kernel<<<gD, blk, 0, stream>>>(hbf, wtv, nullptr, nullptr, nullptr, v, (int)M, D_, D_, 0);

  // Edge bias MLP -> bf16 [B,H,L,L]
  edge_kernel<<<(unsigned)((size_t)B_ * L_ * L_ / 256), blk, 0, stream>>>(cost, We1, be1, We2, be2, ebias);

  // Attention -> ctx bf16
  attn_kernel<<<dim3(L_ / 128, H_, B_), blk, 0, stream>>>(q, k, v, ebias, ctxbf);

  // h = LN1(h_in + ctx @ Wo): fp32 in hbuf, bf16 copy in hbf2
  gemm_kernel<<<gD, blk, 0, stream>>>(ctxbf, wto, nullptr, h_in, hbuf, nullptr, (int)M, D_, D_, 0);
  ln_kernel<<<(unsigned)M, blk, 0, stream>>>(hbuf, g1, bn1, hbf2);

  // ff1 = relu(h @ Wf1 + bf1) -> bf16 ;  out = LN2(h + ff1 @ Wf2 + bf2)
  gemm_kernel<<<gF, blk, 0, stream>>>(hbf2, wtf1, bf1, nullptr, nullptr, ff1, (int)M, D_, FF_, 1);
  gemm_kernel<<<gD, blk, 0, stream>>>(ff1, wtf2, bf2, hbuf, out, nullptr, (int)M, FF_, D_, 0);
  ln_kernel<<<(unsigned)M, blk, 0, stream>>>(out, g2, bn2, nullptr);
}
